// VoxelNet_46729244180826
// MI455X (gfx1250) — compile-verified, matrix operand order checked by reading, not run on hardware
//
#include <hip/hip_runtime.h>

typedef __attribute__((ext_vector_type(16))) _Float16 v16h;
typedef __attribute__((ext_vector_type(8)))  float    v8f;

// Padded grid geometry (shared by all three convs; only #D-planes differs)
#define HP 202
#define WP 178
#define CH 64
#define ROWSTRIDE ((size_t)WP * CH)    // elements per H-row
#define PLANE ((size_t)HP * WP * CH)   // elements per D-plane

// Workspace layout (bytes, all 16B aligned)
static const size_t SZ_DENSE = (size_t)12 * HP * WP * CH * 2;  // 55,228,416
static const size_t SZ_X     = (size_t)5  * HP * WP * CH * 2;  // 23,011,840
static const size_t OFF_DENSE = 0;
static const size_t OFF_X1    = OFF_DENSE + SZ_DENSE;
static const size_t OFF_X2    = OFF_X1 + SZ_X;
static const size_t OFF_WT    = OFF_X2 + SZ_X;                 // 101,252,096
static const size_t SZ_WT     = (size_t)27 * 64 * 64 * 2;      // 221,184

// ---------------- zero fill ----------------
__global__ void zero_ws_kernel(uint4* __restrict__ p, long n16) {
  long i = (long)blockIdx.x * blockDim.x + threadIdx.x;
  if (i < n16) p[i] = make_uint4(0u, 0u, 0u, 0u);
}

// ---------------- scatter voxels (f32 -> fp16, channel-last, +1 halo) ----------------
__global__ void scatter_vox_kernel(const float* __restrict__ feats,
                                   const int* __restrict__ coors,
                                   _Float16* __restrict__ dense, int nvox) {
  int tid = blockIdx.x * blockDim.x + threadIdx.x;
  if (tid >= nvox * 4) return;
  int vox = tid >> 2;
  int c0  = (tid & 3) * 16;
  int d = coors[vox * 4 + 1];
  int h = coors[vox * 4 + 2];
  int w = coors[vox * 4 + 3];
  const float4* f = (const float4*)(feats + (size_t)vox * 64 + c0);
  float4 a = f[0], b = f[1], c = f[2], e = f[3];
  union { uint4 u[2]; _Float16 h16[16]; } o;
  o.h16[0]=(_Float16)a.x;  o.h16[1]=(_Float16)a.y;  o.h16[2]=(_Float16)a.z;  o.h16[3]=(_Float16)a.w;
  o.h16[4]=(_Float16)b.x;  o.h16[5]=(_Float16)b.y;  o.h16[6]=(_Float16)b.z;  o.h16[7]=(_Float16)b.w;
  o.h16[8]=(_Float16)c.x;  o.h16[9]=(_Float16)c.y;  o.h16[10]=(_Float16)c.z; o.h16[11]=(_Float16)c.w;
  o.h16[12]=(_Float16)e.x; o.h16[13]=(_Float16)e.y; o.h16[14]=(_Float16)e.z; o.h16[15]=(_Float16)e.w;
  _Float16* dst = dense + (((size_t)(d + 1) * HP + (h + 1)) * WP + (w + 1)) * CH + c0;
  ((uint4*)dst)[0] = o.u[0];
  ((uint4*)dst)[1] = o.u[1];
}

// ---------------- weight repack: W(oc,ic,k27) f32 -> Wt(k27,oc,ic) fp16 ----------------
__global__ void prep_weights_kernel(const float* __restrict__ w1,
                                    const float* __restrict__ w2,
                                    const float* __restrict__ w3,
                                    _Float16* __restrict__ o1,
                                    _Float16* __restrict__ o2,
                                    _Float16* __restrict__ o3) {
  int tid = blockIdx.x * blockDim.x + threadIdx.x;
  const int per = 27 * 64 * 64;
  if (tid >= 3 * per) return;
  int which = tid / per;
  int rem   = tid % per;
  int k  = rem >> 12;         // / 4096
  int oc = (rem >> 6) & 63;
  int ic = rem & 63;
  const float* src = (which == 0) ? w1 : (which == 1) ? w2 : w3;
  _Float16*    dst = (which == 0) ? o1 : (which == 1) ? o2 : o3;
  dst[((size_t)k * 64 + oc) * 64 + ic] = (_Float16)src[((size_t)oc * 64 + ic) * 27 + k];
}

// ---------------- implicit-GEMM conv via WMMA ----------------
// One wave computes a (2 H-rows x 16 W-positions) x 64-oc tile.
// Each loaded A (weight) fragment is reused by two WMMAs (one per row):
// per 32-K chunk: 4 B-loads + 8 A-loads (b128) -> 8 WMMAs.
// OUT_F32 = 0: fused scale/bias/relu -> fp16 padded channel-last output
// OUT_F32 = 1: fused scale/bias/relu -> f32 output in (oc*2+d, 200, 176) layout
template <int OUT_F32>
__global__ __launch_bounds__(128) void conv_wmma_kernel(
    const _Float16* __restrict__ in,   // padded (D, 202, 178, 64) fp16
    const _Float16* __restrict__ wt,   // (27, 64, 64) fp16
    const float* __restrict__ scale,
    const float* __restrict__ bias,
    void* __restrict__ outp,
    int sd,          // D stride of the conv (input plane index = dout*sd + kd)
    int d_off_out,   // D halo offset when storing fp16 output
    int ntiles) {
  int wave = blockIdx.x * (blockDim.x >> 5) + (threadIdx.x >> 5);
  if (wave >= ntiles) return;           // wave-uniform; EXEC stays all-ones
  int lane = threadIdx.x & 31;
  int n    = lane & 15;                 // N (position) / M (oc) index within tile
  int half = lane >> 4;                 // K-half selector per ISA layout

  int wtile = wave % 11;
  int tmp   = wave / 11;
  int hpair = tmp % 100;                // 2 consecutive H rows per wave
  int dout  = tmp / 100;
  int hout  = hpair * 2;
  int wbase = wtile * 16;

  v8f acc[4][2];
#pragma unroll
  for (int t = 0; t < 4; ++t) {
    acc[t][0] = (v8f){};
    acc[t][1] = (v8f){};
  }

  // Base input pointer for this lane's B columns (position wbase+n, rows hout / hout+1)
  const _Float16* inbase = in + (size_t)(dout * sd) * PLANE
                              + (size_t)hout * ROWSTRIDE
                              + (size_t)(wbase + n) * CH;

  for (int kd = 0; kd < 3; ++kd) {
    for (int kh = 0; kh < 3; ++kh) {
      for (int kw = 0; kw < 3; ++kw) {
        int k = (kd * 3 + kh) * 3 + kw;
        const _Float16* ip0 = inbase + (size_t)kd * PLANE + (size_t)kh * ROWSTRIDE + kw * CH;
        const _Float16* ip1 = ip0 + ROWSTRIDE;
        const _Float16* wp = wt + (size_t)k * 64 * 64;
#pragma unroll
        for (int chunk = 0; chunk < 64; chunk += 32) {
          // B fragments: 32 K-values (ic chunk) x 16 positions, for both rows
          union { v16h v; uint4 u[2]; } bf0, bf1;
          bf0.u[0] = *(const uint4*)(ip0 + chunk + half * 8);
          bf0.u[1] = *(const uint4*)(ip0 + chunk + 16 + half * 8);
          bf1.u[0] = *(const uint4*)(ip1 + chunk + half * 8);
          bf1.u[1] = *(const uint4*)(ip1 + chunk + 16 + half * 8);
          // A fragments: this lane's row oc = t*16 + n, same K split
          const _Float16* wl = wp + (size_t)n * 64 + chunk + half * 8;
#pragma unroll
          for (int t = 0; t < 4; ++t) {
            union { v16h v; uint4 u[2]; } af;
            af.u[0] = *(const uint4*)(wl + t * 1024);
            af.u[1] = *(const uint4*)(wl + t * 1024 + 16);
            acc[t][0] = __builtin_amdgcn_wmma_f32_16x16x32_f16(
                false, af.v, false, bf0.v, (short)0, acc[t][0], false, false);
            acc[t][1] = __builtin_amdgcn_wmma_f32_16x16x32_f16(
                false, af.v, false, bf1.v, (short)0, acc[t][1], false, false);
          }
        }
      }
    }
  }

  if (OUT_F32 == 0) {
    _Float16* out = (_Float16*)outp;
#pragma unroll
    for (int r = 0; r < 2; ++r) {
      size_t obase = (((size_t)(dout + d_off_out) * HP + (hout + r + 1)) * WP
                      + (wbase + n + 1)) * CH + half * 8;
#pragma unroll
      for (int t = 0; t < 4; ++t) {
        float4 s0 = *(const float4*)(scale + t * 16 + half * 8);
        float4 s1 = *(const float4*)(scale + t * 16 + half * 8 + 4);
        float4 b0 = *(const float4*)(bias + t * 16 + half * 8);
        float4 b1 = *(const float4*)(bias + t * 16 + half * 8 + 4);
        float sc[8] = {s0.x, s0.y, s0.z, s0.w, s1.x, s1.y, s1.z, s1.w};
        float bi[8] = {b0.x, b0.y, b0.z, b0.w, b1.x, b1.y, b1.z, b1.w};
        union { uint4 u; _Float16 h[8]; } o;
#pragma unroll
        for (int v = 0; v < 8; ++v) {
          float val = acc[t][r][v] * sc[v] + bi[v];
          val = val > 0.0f ? val : 0.0f;
          o.h[v] = (_Float16)val;
        }
        *(uint4*)(out + obase + t * 16) = o.u;
      }
    }
  } else {
    float* out = (float*)outp;   // (oc*2 + dout, 200, 176)
#pragma unroll
    for (int t = 0; t < 4; ++t) {
      float4 s0 = *(const float4*)(scale + t * 16 + half * 8);
      float4 s1 = *(const float4*)(scale + t * 16 + half * 8 + 4);
      float4 b0 = *(const float4*)(bias + t * 16 + half * 8);
      float4 b1 = *(const float4*)(bias + t * 16 + half * 8 + 4);
      float sc[8] = {s0.x, s0.y, s0.z, s0.w, s1.x, s1.y, s1.z, s1.w};
      float bi[8] = {b0.x, b0.y, b0.z, b0.w, b1.x, b1.y, b1.z, b1.w};
#pragma unroll
      for (int r = 0; r < 2; ++r) {
#pragma unroll
        for (int v = 0; v < 8; ++v) {
          int oc = t * 16 + half * 8 + v;
          float val = acc[t][r][v] * sc[v] + bi[v];
          val = val > 0.0f ? val : 0.0f;
          out[(((size_t)(oc * 2 + dout)) * 200 + (hout + r)) * 176 + (wbase + n)] = val;
        }
      }
    }
  }
}

extern "C" void kernel_launch(void* const* d_in, const int* in_sizes, int n_in,
                              void* d_out, int out_size, void* d_ws, size_t ws_size,
                              hipStream_t stream) {
  const float* feats  = (const float*)d_in[0];
  const int*   coors  = (const int*)d_in[1];
  // d_in[2] = batch_size (always 1) — unused
  const float* W1     = (const float*)d_in[3];
  const float* scale1 = (const float*)d_in[4];
  const float* bias1  = (const float*)d_in[5];
  const float* W2     = (const float*)d_in[6];
  const float* scale2 = (const float*)d_in[7];
  const float* bias2  = (const float*)d_in[8];
  const float* W3     = (const float*)d_in[9];
  const float* scale3 = (const float*)d_in[10];
  const float* bias3  = (const float*)d_in[11];

  char* ws = (char*)d_ws;
  _Float16* dense = (_Float16*)(ws + OFF_DENSE);
  _Float16* x1    = (_Float16*)(ws + OFF_X1);
  _Float16* x2    = (_Float16*)(ws + OFF_X2);
  _Float16* wt1   = (_Float16*)(ws + OFF_WT);
  _Float16* wt2   = (_Float16*)(ws + OFF_WT + SZ_WT);
  _Float16* wt3   = (_Float16*)(ws + OFF_WT + 2 * SZ_WT);

  // 1) zero dense grid + both padded intermediates (halos become implicit zeros)
  long n16 = (long)(OFF_WT / 16);
  zero_ws_kernel<<<(int)((n16 + 255) / 256), 256, 0, stream>>>((uint4*)ws, n16);

  // 2) scatter voxel features (f32 -> fp16, channel-last)
  int nvox = in_sizes[0] / 64;
  scatter_vox_kernel<<<(nvox * 4 + 255) / 256, 256, 0, stream>>>(feats, coors, dense, nvox);

  // 3) repack all three weight tensors
  prep_weights_kernel<<<(3 * 27 * 64 * 64) / 256, 256, 0, stream>>>(W1, W2, W3, wt1, wt2, wt3);

  // 4) conv1: stride (2,1,1), pad (1,1,1): dense(12 planes) -> x1 (5 planes, no D halo)
  int nt1 = 5 * 100 * 11;
  conv_wmma_kernel<0><<<nt1 / 4, 128, 0, stream>>>(dense, wt1, scale1, bias1, x1, 2, 0, nt1);

  // 5) conv2: stride 1, pad (0,1,1): x1 (5 planes) -> x2 interior (planes 1..3 of 5)
  int nt2 = 3 * 100 * 11;
  conv_wmma_kernel<0><<<nt2 / 4, 128, 0, stream>>>(x1, wt2, scale2, bias2, x2, 1, 1, nt2);

  // 6) conv3: stride (2,1,1), pad (1,1,1): x2 (5 planes incl halo) -> f32 reshaped output
  int nt3 = 2 * 100 * 11;
  conv_wmma_kernel<1><<<nt3 / 4, 128, 0, stream>>>(x2, wt3, scale3, bias3, d_out, 2, 0, nt3);
}
